// Graph_GAT_75196287418587
// MI455X (gfx1250) — matrix operand order repaired and checked by reading, not compiled, not run on hardware
//
#include <hip/hip_runtime.h>
#include <hip/hip_bf16.h>
#include <math.h>

typedef __attribute__((ext_vector_type(16))) _Float16 v16h;
typedef __attribute__((ext_vector_type(8)))  float    v8f;
typedef unsigned int u32x4 __attribute__((ext_vector_type(4)));
typedef int          i32x8 __attribute__((ext_vector_type(8)));
typedef int          i32x4 __attribute__((ext_vector_type(4)));

#define TPB 256
static inline dim3 grd(long n) { return dim3((unsigned)((n + TPB - 1) / TPB)); }

// ---------------------------------------------------------------------------
// Fill / convert kernels
// ---------------------------------------------------------------------------
__global__ void k_fill_f32(float* p, float v, long n) {
  long i = (long)blockIdx.x * blockDim.x + threadIdx.x;
  if (i < n) p[i] = v;
}
__global__ void k_fill_u32(unsigned* p, unsigned v, long n) {
  long i = (long)blockIdx.x * blockDim.x + threadIdx.x;
  if (i < n) p[i] = v;
}
__global__ void k_cvt_f16(const float* __restrict__ src, _Float16* __restrict__ dst, long n) {
  long i = ((long)blockIdx.x * blockDim.x + threadIdx.x) * 2;
  if (i + 1 < n) {
    dst[i]     = (_Float16)src[i];
    dst[i + 1] = (_Float16)src[i + 1];
  } else if (i < n) {
    dst[i] = (_Float16)src[i];
  }
}

// ---------------------------------------------------------------------------
// Tensor Data Mover: 2-D f16 tile load Global -> LDS.
// D# per CDNA5 ISA group0/group1 layout. data_size = 2 bytes.
// rem0/rem1: remaining tensor extent from tile start (OOB reads return 0).
// pad_code:  pad interval = 2^pad_code x 8 bytes; pad_amount = 2 DWORDs
//            => LDS row stride = tile0 + 4 halves.
// ---------------------------------------------------------------------------
__device__ inline void tdm_load_2d(unsigned lds_off, const _Float16* gptr,
                                   int rem0, int rem1, int tile0, int tile1,
                                   int stride0, int pad_code) {
  unsigned long long ga = (unsigned long long)(void*)gptr;
  u32x4 g0;
  g0[0] = 1u;                                     // count=1 (valid user D#)
  g0[1] = lds_off;                                // lds_addr (bytes)
  g0[2] = (unsigned)ga;                           // global_addr[31:0]
  g0[3] = (unsigned)((ga >> 32) & 0x01ffffffull)  // global_addr[56:32]
          | (2u << 30);                           // type = 2 ("image")
  i32x8 g1;
  g1[0] = (1 << 16)            // data_size = 2 bytes
        | (1 << 20)            // pad_enable
        | (pad_code << 22)     // pad_interval
        | (1 << 25);           // pad_amount = 2 DWORDs (4 halves)
  g1[1] = (int)(((unsigned)rem0 & 0xffffu) << 16);                   // dim0[15:0]
  g1[2] = (int)((((unsigned)rem0 >> 16) & 0xffffu) |
                (((unsigned)rem1 & 0xffffu) << 16));                 // dim0 hi, dim1 lo
  g1[3] = (int)((((unsigned)rem1 >> 16) & 0xffffu) |
                (((unsigned)tile0 & 0xffffu) << 16));                // dim1 hi, tile_dim0
  g1[4] = tile1 & 0xffff;                                            // tile_dim1 (tile_dim2=0)
  g1[5] = stride0;                                                   // dim0_stride[31:0]
  g1[6] = 0;
  g1[7] = 0;
  i32x4 g2 = {0, 0, 0, 0};
  i32x4 g3 = {0, 0, 0, 0};
#if __clang_major__ >= 23
  i32x8 g4 = {0, 0, 0, 0, 0, 0, 0, 0};
  __builtin_amdgcn_tensor_load_to_lds(g0, g1, g2, g3, g4, 0);
#else
  __builtin_amdgcn_tensor_load_to_lds(g0, g1, g2, g3, 0);
#endif
}

// ---------------------------------------------------------------------------
// WMMA GEMM: C[M,N] = act(A16[M,K] @ W16[K,N] + bias); act 0=none 1=relu 2=tanh
// Block = 256 threads (8 waves). Block tile = 32 rows x 128 cols, K chunk 64.
// Wave w owns cols [w*16, w*16+16) and both 16-row tiles -> 4 WMMA / chunk.
// f16 tiles staged via double-buffered TDM (prefetch next chunk while
// computing current; tensorcnt<=2 waits for current since TDM is in-order).
// ---------------------------------------------------------------------------
#define A_STR 68    // 64 + 4 pad halves
#define B_STR 132   // 128 + 4 pad halves

__global__ __launch_bounds__(256) void k_gemm_wmma(
    const _Float16* __restrict__ A, const _Float16* __restrict__ W,
    const float* __restrict__ bias, float* __restrict__ C,
    int M, int K, int N, int act) {
  __shared__ _Float16 Asf[2][32 * A_STR];   // 2 x 4.25 KB
  __shared__ _Float16 Bsf[2][64 * B_STR];   // 2 x 16.5 KB

  const int tid  = threadIdx.x;
  const int wave = tid >> 5;
  const int lane = tid & 31;
  const int hi   = lane >> 4;
  const int ln   = lane & 15;
  const int rowBase = blockIdx.y * 32;
  const int colBase = blockIdx.x * 128;
  const int bcol = wave * 16 + ln;

  const int nChunks = (K + 63) / 64;

  if (wave == 0) {
    // chunk 0 into buffer 0
    tdm_load_2d((unsigned)(unsigned long long)(void*)&Asf[0][0],
                A + (long)rowBase * K, K, M - rowBase, 64, 32, K, 4);
    tdm_load_2d((unsigned)(unsigned long long)(void*)&Bsf[0][0],
                W + colBase, N - colBase, K, 128, 64, N, 5);
  }

  v8f acc0 = {}, acc1 = {};
  for (int ci = 0; ci < nChunks; ++ci) {
    const int b = ci & 1;
    if (wave == 0) {
      if (ci + 1 < nChunks) {
        const int kn = (ci + 1) * 64;
        tdm_load_2d((unsigned)(unsigned long long)(void*)&Asf[b ^ 1][0],
                    A + (long)rowBase * K + kn, K - kn, M - rowBase, 64, 32, K, 4);
        tdm_load_2d((unsigned)(unsigned long long)(void*)&Bsf[b ^ 1][0],
                    W + (long)kn * N + colBase, N - colBase, K - kn, 128, 64, N, 5);
        __builtin_amdgcn_s_wait_tensorcnt(2);   // current chunk complete
      } else {
        __builtin_amdgcn_s_wait_tensorcnt(0);
      }
    }
    __syncthreads();   // buf b ready; prior buf no longer being overwritten

    const _Float16* Ab = Asf[b];
    const _Float16* Bb = Bsf[b];
    const int kb = hi * 16;
#pragma unroll
    for (int q = 0; q < 2; ++q) {     // two 32-wide K sub-chunks
      const int ks = q * 32;
      v16h af0, af1, bf;
#pragma unroll
      for (int e = 0; e < 8; ++e) {
        af0[e]     = Ab[ln * A_STR + ks + kb + e];
        af0[e + 8] = Ab[ln * A_STR + ks + 8 + kb + e];
        af1[e]     = Ab[(16 + ln) * A_STR + ks + kb + e];
        af1[e + 8] = Ab[(16 + ln) * A_STR + ks + 8 + kb + e];
      }
#pragma unroll
      for (int e = 0; e < 16; ++e)
        bf[e] = Bb[(ks + hi * 16 + e) * B_STR + bcol];

      acc0 = __builtin_amdgcn_wmma_f32_16x16x32_f16(
          false, af0, false, bf, (short)0, acc0, false, false);
      acc1 = __builtin_amdgcn_wmma_f32_16x16x32_f16(
          false, af1, false, bf, (short)0, acc1, false, false);
    }
    __syncthreads();   // done reading buf b before it is overwritten
  }

  // C layout: VGPR r -> M = r + 8*hi, N = lane&15
  const int gc = colBase + bcol;
#pragma unroll
  for (int t = 0; t < 2; ++t) {
    const v8f& acc = t ? acc1 : acc0;
#pragma unroll
    for (int r = 0; r < 8; ++r) {
      int gr = rowBase + t * 16 + hi * 8 + r;
      if (gr < M && gc < N) {
        float v = acc[r];
        if (bias) v += bias[gc];
        if (act == 1) v = v > 0.0f ? v : 0.0f;
        else if (act == 2) v = tanhf(v);
        C[(long)gr * N + gc] = v;
      }
    }
  }
}

static void gemm_launch(const float* A, const float* W, const float* bias,
                        float* C, int M, int K, int N, int act,
                        _Float16* a16, _Float16* w16, hipStream_t st) {
  long na = (long)M * K, nw = (long)K * N;
  k_cvt_f16<<<grd((na + 1) / 2), TPB, 0, st>>>(A, a16, na);
  k_cvt_f16<<<grd((nw + 1) / 2), TPB, 0, st>>>(W, w16, nw);
  dim3 grid((N + 127) / 128, (M + 31) / 32);
  k_gemm_wmma<<<grid, TPB, 0, st>>>(a16, w16, bias, C, M, K, N, act);
}

// ---------------------------------------------------------------------------
// Float atomic max via order-preserving uint encoding
// ---------------------------------------------------------------------------
__device__ inline unsigned f2ord(float f) {
  unsigned u = __float_as_uint(f);
  return (u & 0x80000000u) ? ~u : (u | 0x80000000u);
}
__device__ inline float ord2f(unsigned u) {
  return (u & 0x80000000u) ? __uint_as_float(u & 0x7fffffffu)
                           : __uint_as_float(~u);
}
#define ORD_NEG_INF 0x007FFFFFu   // f2ord(-inf)

// ---------------------------------------------------------------------------
// GAT kernels (HEADS = 2)
// ---------------------------------------------------------------------------
__global__ void k_gat_scores(const float* __restrict__ h2,
                             const float* __restrict__ a_src,
                             const float* __restrict__ a_dst,
                             float* __restrict__ ssrc, float* __restrict__ sdst,
                             int Nn, int Cc) {
  long i = (long)blockIdx.x * blockDim.x + threadIdx.x;
  if (i >= (long)Nn * 2) return;
  int n = (int)(i >> 1), h = (int)(i & 1);
  const float* hp = h2 + (long)n * 2 * Cc + (long)h * Cc;
  const float* as = a_src + h * Cc;
  const float* ad = a_dst + h * Cc;
  float s1 = 0.0f, s2 = 0.0f;
  for (int c = 0; c < Cc; ++c) { float v = hp[c]; s1 += v * as[c]; s2 += v * ad[c]; }
  ssrc[i] = s1; sdst[i] = s2;
}

__global__ void k_gat_logit_max(const int* __restrict__ esrc, const int* __restrict__ edst,
                                int E, int Nn, const float* __restrict__ ssrc,
                                const float* __restrict__ sdst,
                                float* __restrict__ logits, unsigned* __restrict__ mord) {
  long i = (long)blockIdx.x * blockDim.x + threadIdx.x;
  long total = (long)(E + Nn) * 2;
  if (i >= total) return;
  int e = (int)(i >> 1), h = (int)(i & 1);
  int s, d;
  if (e < E) { s = esrc[e]; d = edst[e]; } else { s = d = e - E; }
  float l = ssrc[s * 2 + h] + sdst[d * 2 + h];
  l = l > 0.0f ? l : 0.2f * l;   // leaky_relu
  logits[i] = l;
  atomicMax(&mord[d * 2 + h], f2ord(l));
}

__global__ void k_gat_exp_sum(const int* __restrict__ edst, int E, int Nn,
                              float* __restrict__ logits,
                              const unsigned* __restrict__ mord,
                              float* __restrict__ ssum) {
  long i = (long)blockIdx.x * blockDim.x + threadIdx.x;
  long total = (long)(E + Nn) * 2;
  if (i >= total) return;
  int e = (int)(i >> 1), h = (int)(i & 1);
  int d = (e < E) ? edst[e] : e - E;
  float m = ord2f(mord[d * 2 + h]);
  float ex = __expf(logits[i] - m);
  logits[i] = ex;
  atomicAdd(&ssum[d * 2 + h], ex);
}

__global__ void k_gat_agg(const int* __restrict__ esrc, const int* __restrict__ edst,
                          int E, int Nn, const float* __restrict__ alpha,
                          const float* __restrict__ ssum,
                          const float* __restrict__ h2, float* __restrict__ hm, int Cc) {
  long e = (long)blockIdx.x * blockDim.x + threadIdx.x;
  if (e >= (long)(E + Nn)) return;
  int s, d;
  if (e < E) { s = esrc[e]; d = edst[e]; } else { s = d = (int)e - E; }
  float* op = hm + (long)d * Cc;
#pragma unroll
  for (int h = 0; h < 2; ++h) {
    float a = alpha[e * 2 + h] / ssum[d * 2 + h] * 0.5f;  // 0.5 = head mean
    const float* hp = h2 + (long)s * 2 * Cc + (long)h * Cc;
    for (int c = 0; c < Cc; ++c) atomicAdd(&op[c], a * hp[c]);
  }
}

__global__ void k_bias_act(const float* __restrict__ src, const float* __restrict__ bias,
                           float* __restrict__ dst, long total, int Cc, int act) {
  long i = (long)blockIdx.x * blockDim.x + threadIdx.x;
  if (i >= total) return;
  int c = (int)(i % Cc);
  float v = src[i] + (bias ? bias[c] : 0.0f);
  if (act == 1) v = v > 0.0f ? v : 0.0f;
  dst[i] = v;
}

// ---------------------------------------------------------------------------
// GCN kernels
// ---------------------------------------------------------------------------
__global__ void k_gcn_deg(const int* __restrict__ edst, const float* __restrict__ ew,
                          int E, int Nn, float* __restrict__ deg) {
  long e = (long)blockIdx.x * blockDim.x + threadIdx.x;
  if (e >= (long)(E + Nn)) return;
  int d = (e < E) ? edst[e] : (int)e - E;
  float w = (e < E) ? ew[e] : 1.0f;
  atomicAdd(&deg[d], w);
}
__global__ void k_gcn_dinv(float* deg, int Nn) {
  long i = (long)blockIdx.x * blockDim.x + threadIdx.x;
  if (i >= Nn) return;
  float d = deg[i];
  deg[i] = d > 0.0f ? rsqrtf(fmaxf(d, 1e-12f)) : 0.0f;
}
__global__ void k_gcn_agg(const int* __restrict__ esrc, const int* __restrict__ edst,
                          const float* __restrict__ ew, int E, int Nn,
                          const float* __restrict__ dinv, const float* __restrict__ h,
                          float* __restrict__ hm, int Cc) {
  long e = (long)blockIdx.x * blockDim.x + threadIdx.x;
  if (e >= (long)(E + Nn)) return;
  int s, d; float w;
  if (e < E) { s = esrc[e]; d = edst[e]; w = ew[e]; }
  else       { s = d = (int)e - E; w = 1.0f; }
  float norm = dinv[s] * w * dinv[d];
  const float* hp = h + (long)s * Cc;
  float* op = hm + (long)d * Cc;
  for (int c = 0; c < Cc; ++c) atomicAdd(&op[c], norm * hp[c]);
}

// ---------------------------------------------------------------------------
// Gate, pool, attention, head
// ---------------------------------------------------------------------------
__global__ void k_gate(const float* __restrict__ g1o, const float* __restrict__ g2o,
                       const float* __restrict__ gb, const float* __restrict__ h,
                       float* __restrict__ x, long total, int Cc) {
  long i = (long)blockIdx.x * blockDim.x + threadIdx.x;
  if (i >= total) return;
  int c = (int)(i % Cc);
  float z = 1.0f / (1.0f + __expf(-(g1o[i] + g2o[i] + gb[c])));
  x[i] = z * h[i] + (1.0f - z) * x[i];
}

__global__ void k_pool_sum(const float* __restrict__ x, const int* __restrict__ batch,
                           float* __restrict__ gsum, float* __restrict__ cnt,
                           int Nn, int Cc) {
  long n = (long)blockIdx.x * blockDim.x + threadIdx.x;
  if (n >= Nn) return;
  int b = batch[n];
  atomicAdd(&cnt[b], 1.0f);
  const float* xp = x + n * Cc;
  float* gp = gsum + (long)b * Cc;
  for (int c = 0; c < Cc; ++c) atomicAdd(&gp[c], xp[c]);
}
__global__ void k_pool_div(float* __restrict__ g, const float* __restrict__ cnt,
                           int Cc, long total) {
  long i = (long)blockIdx.x * blockDim.x + threadIdx.x;
  if (i >= total) return;
  int b = (int)(i / Cc);
  g[i] /= fmaxf(cnt[b], 1.0f);
}

__global__ void k_att_dot(const float* __restrict__ t64, const float* __restrict__ l2w,
                          float* __restrict__ scores, int Bb, int col) {
  long b = (long)blockIdx.x * blockDim.x + threadIdx.x;
  if (b >= Bb) return;
  float s = 0.0f;
  const float* tp = t64 + b * 64;
  for (int k = 0; k < 64; ++k) s += tp[k] * l2w[k];
  scores[b * 3 + col] = s;
}

__global__ void k_att_emb(const float* __restrict__ scores, const float* __restrict__ ex,
                          const float* __restrict__ ext, const float* __restrict__ exq,
                          float* __restrict__ emb, int Bb) {
  long i = (long)blockIdx.x * blockDim.x + threadIdx.x;
  if (i >= (long)Bb * 128) return;
  int b = (int)(i >> 7), c = (int)(i & 127);
  float s0 = scores[b * 3 + 0], s1 = scores[b * 3 + 1], s2 = scores[b * 3 + 2];
  float m = fmaxf(s0, fmaxf(s1, s2));
  float e0 = __expf(s0 - m), e1 = __expf(s1 - m), e2 = __expf(s2 - m);
  float inv = 1.0f / (e0 + e1 + e2);
  emb[(long)b * 384 + c]       = e0 * inv * ex[i];
  emb[(long)b * 384 + 128 + c] = e1 * inv * ext[i];
  emb[(long)b * 384 + 256 + c] = e2 * inv * exq[i];
}

// ---------------------------------------------------------------------------
// Host orchestration
// ---------------------------------------------------------------------------
struct Gat { const float *W, *adst, *asrc, *b; };
struct BranchP {
  Gat c0, c1, c2;   // c0.adst/asrc == nullptr means GCN layer 0
  const float *f1W, *f1b, *f2W, *f2b, *g1W, *g1b, *g2W, *g2b, *gb;
};
struct Scratch {
  float *xbuf, *h2, *hm, *g1o, *g2o, *ssrc, *sdst, *ssum, *alpha, *deg, *gpool, *cnt, *f1o;
  unsigned* mord;
  _Float16 *a16, *w16;
};

static void run_gat_layer(const float* xin, int Nn, int fin, int Cc,
                          const int* esrc, const int* edst, int E, const Gat& g,
                          Scratch& sc, float* xout, int act, hipStream_t st) {
  gemm_launch(xin, g.W, nullptr, sc.h2, Nn, fin, 2 * Cc, 0, sc.a16, sc.w16, st);
  long nh = (long)Nn * 2;
  k_gat_scores<<<grd(nh), TPB, 0, st>>>(sc.h2, g.asrc, g.adst, sc.ssrc, sc.sdst, Nn, Cc);
  k_fill_u32<<<grd(nh), TPB, 0, st>>>(sc.mord, ORD_NEG_INF, nh);
  k_fill_f32<<<grd(nh), TPB, 0, st>>>(sc.ssum, 0.0f, nh);
  long esl2 = (long)(E + Nn) * 2;
  k_gat_logit_max<<<grd(esl2), TPB, 0, st>>>(esrc, edst, E, Nn, sc.ssrc, sc.sdst, sc.alpha, sc.mord);
  k_gat_exp_sum<<<grd(esl2), TPB, 0, st>>>(edst, E, Nn, sc.alpha, sc.mord, sc.ssum);
  long nc = (long)Nn * Cc;
  k_fill_f32<<<grd(nc), TPB, 0, st>>>(sc.hm, 0.0f, nc);
  k_gat_agg<<<grd((long)(E + Nn)), TPB, 0, st>>>(esrc, edst, E, Nn, sc.alpha, sc.ssum, sc.h2, sc.hm, Cc);
  k_bias_act<<<grd(nc), TPB, 0, st>>>(sc.hm, g.b, xout, nc, Cc, act);
}

static void run_branch(const float* xin, int Nn, int fin, int d,
                       const int* ei, int E, const int* batch, const float* ew,
                       const BranchP& bp, Scratch& sc, float* emb_out, hipStream_t st) {
  const int* esrc = ei;
  const int* edst = ei + E;
  long nd = (long)Nn * d;

  // ---- layer 0 ----
  if (ew) {  // GCN
    gemm_launch(xin, bp.c0.W, nullptr, sc.g1o, Nn, fin, d, 0, sc.a16, sc.w16, st);
    k_fill_f32<<<grd((long)Nn), TPB, 0, st>>>(sc.deg, 0.0f, Nn);
    k_gcn_deg<<<grd((long)(E + Nn)), TPB, 0, st>>>(edst, ew, E, Nn, sc.deg);
    k_gcn_dinv<<<grd((long)Nn), TPB, 0, st>>>(sc.deg, Nn);
    k_fill_f32<<<grd(nd), TPB, 0, st>>>(sc.hm, 0.0f, nd);
    k_gcn_agg<<<grd((long)(E + Nn)), TPB, 0, st>>>(esrc, edst, ew, E, Nn, sc.deg, sc.g1o, sc.hm, d);
    k_bias_act<<<grd(nd), TPB, 0, st>>>(sc.hm, bp.c0.b, sc.xbuf, nd, d, 1);
  } else {   // GAT
    run_gat_layer(xin, Nn, fin, d, esrc, edst, E, bp.c0, sc, sc.xbuf, 1, st);
  }

  // ---- layers 1 & 2 with gated residual ----
  for (int layer = 1; layer <= 2; ++layer) {
    const Gat& g = (layer == 1) ? bp.c1 : bp.c2;
    int act = (layer == 1) ? 1 : 0;
    run_gat_layer(sc.xbuf, Nn, d, d, esrc, edst, E, g, sc, sc.hm, act, st);
    gemm_launch(sc.hm, bp.g1W, bp.g1b, sc.g1o, Nn, d, d, 0, sc.a16, sc.w16, st);
    gemm_launch(sc.xbuf, bp.g2W, bp.g2b, sc.g2o, Nn, d, d, 0, sc.a16, sc.w16, st);
    k_gate<<<grd(nd), TPB, 0, st>>>(sc.g1o, sc.g2o, bp.gb, sc.hm, sc.xbuf, nd, d);
  }

  // ---- global mean pool + branch FC ----
  long bd = 256L * d;
  k_fill_f32<<<grd(bd), TPB, 0, st>>>(sc.gpool, 0.0f, bd);
  k_fill_f32<<<grd(256L), TPB, 0, st>>>(sc.cnt, 0.0f, 256);
  k_pool_sum<<<grd((long)Nn), TPB, 0, st>>>(sc.xbuf, batch, sc.gpool, sc.cnt, Nn, d);
  k_pool_div<<<grd(bd), TPB, 0, st>>>(sc.gpool, sc.cnt, d, bd);
  gemm_launch(sc.gpool, bp.f1W, bp.f1b, sc.f1o, 256, d, 1024, 1, sc.a16, sc.w16, st);
  gemm_launch(sc.f1o, bp.f2W, bp.f2b, emb_out, 256, 1024, 128, 0, sc.a16, sc.w16, st);
}

extern "C" void kernel_launch(void* const* d_in, const int* in_sizes, int n_in,
                              void* d_out, int out_size, void* d_ws, size_t ws_size,
                              hipStream_t stream) {
  (void)n_in; (void)out_size; (void)ws_size;
  // -------- data inputs --------
  const float* mol_x    = (const float*)d_in[0];
  const int*   mol_ei   = (const int*)d_in[1];
  const int*   mol_bat  = (const int*)d_in[2];
  const float* pro_x    = (const float*)d_in[3];
  const int*   pro_ei   = (const int*)d_in[4];
  const float* pro_ew   = (const float*)d_in[5];
  const int*   pro_bat  = (const int*)d_in[6];
  const float* clq_x    = (const float*)d_in[7];
  const int*   clq_ei   = (const int*)d_in[8];
  const int*   clq_bat  = (const int*)d_in[9];

  const int molN = in_sizes[0] / 78, molE = in_sizes[1] / 2, molD = 78 * 4;
  const int proN = in_sizes[3] / 33, proE = in_sizes[4] / 2, proD = 33 * 4;
  const int clqN = in_sizes[7] / 92, clqE = in_sizes[8] / 2, clqD = 92 * 4;

  // -------- params (pytree: sorted dict keys; GAT leaf order W, a_dst, a_src, b) --------
  const float* P[96];
  for (int i = 10; i < n_in && i < 96; ++i) P[i] = (const float*)d_in[i];
  const float *attx_l1W = P[10], *attx_l1b = P[11], *attx_l2W = P[12];
  const float *attq_l1W = P[13], *attq_l1b = P[14], *attq_l2W = P[15];
  const float *attt_l1W = P[16], *attt_l1b = P[17], *attt_l2W = P[18];

  BranchP clq{};
  clq.c0 = {P[19], P[20], P[21], P[22]};
  clq.c1 = {P[23], P[24], P[25], P[26]};
  clq.c2 = {P[27], P[28], P[29], P[30]};
  clq.f1W = P[31]; clq.f1b = P[32]; clq.f2W = P[33]; clq.f2b = P[34];
  clq.g1W = P[35]; clq.g1b = P[36]; clq.g2W = P[37]; clq.g2b = P[38]; clq.gb = P[39];

  const float *fc1W = P[40], *fc1b = P[41], *fc2W = P[42], *fc2b = P[43];

  BranchP mol{};
  mol.c0 = {P[44], P[45], P[46], P[47]};
  mol.c1 = {P[48], P[49], P[50], P[51]};
  mol.c2 = {P[52], P[53], P[54], P[55]};
  mol.f1W = P[56]; mol.f1b = P[57]; mol.f2W = P[58]; mol.f2b = P[59];
  mol.g1W = P[60]; mol.g1b = P[61]; mol.g2W = P[62]; mol.g2b = P[63]; mol.gb = P[64];

  const float *outW = P[65], *outb = P[66];

  BranchP pro{};
  pro.c0 = {P[67], nullptr, nullptr, P[68]};   // GCN: W, b
  pro.c1 = {P[69], P[70], P[71], P[72]};
  pro.c2 = {P[73], P[74], P[75], P[76]};
  pro.f1W = P[77]; pro.f1b = P[78]; pro.f2W = P[79]; pro.f2b = P[80];
  pro.g1W = P[81]; pro.g1b = P[82]; pro.g2W = P[83]; pro.g2b = P[84]; pro.gb = P[85];

  // -------- workspace bump allocator --------
  char* wsB = (char*)d_ws;
  size_t off = 0;
  auto allocB = [&](size_t bytes) -> void* {
    void* p = (void*)(wsB + off);
    off = (off + bytes + 255) & ~(size_t)255;
    return p;
  };
  auto alloc = [&](long nfloats) -> float* { return (float*)allocB((size_t)nfloats * 4); };
  auto lmax3 = [](long a, long b, long c) { long m = a > b ? a : b; return m > c ? m : c; };

  long nd   = lmax3((long)molN * molD, (long)proN * proD, (long)clqN * clqD);
  long n2d  = lmax3((long)molN * 2 * molD, (long)proN * 2 * proD, (long)clqN * 2 * clqD);
  long nh   = lmax3((long)molN, (long)proN, (long)clqN) * 2;
  long eslh = lmax3((long)(molE + molN), (long)(proE + proN), (long)(clqE + clqN)) * 2;
  int  dmax = molD > proD ? (molD > clqD ? molD : clqD) : (proD > clqD ? proD : clqD);

  Scratch sc{};
  sc.xbuf  = alloc(nd);
  sc.h2    = alloc(n2d);
  sc.hm    = alloc(nd);
  sc.g1o   = alloc(nd);
  sc.g2o   = alloc(nd);
  sc.ssrc  = alloc(nh);
  sc.sdst  = alloc(nh);
  sc.ssum  = alloc(nh);
  sc.mord  = (unsigned*)alloc(nh);
  sc.alpha = alloc(eslh);
  sc.deg   = alloc(lmax3(molN, proN, clqN));
  sc.gpool = alloc(256L * dmax);
  sc.cnt   = alloc(256);
  sc.f1o   = alloc(256L * 1024);
  sc.a16   = (_Float16*)allocB((size_t)nd * 2);        // max M*K halves
  sc.w16   = (_Float16*)allocB((size_t)(1 << 20) * 2); // max K*N halves (fc2: 512K)
  float* embx   = alloc(256L * 128);
  float* embxt  = alloc(256L * 128);
  float* embxq  = alloc(256L * 128);
  float* t64    = alloc(256L * 64);
  float* scores = alloc(256L * 3);
  float* embcat = alloc(256L * 384);
  float* fc1o   = alloc(256L * 1024);
  float* fc2o   = alloc(256L * 512);

  // -------- three branches (sequential, scratch reused) --------
  run_branch(mol_x, molN, 78, molD, mol_ei, molE, mol_bat, nullptr, mol, sc, embx,  stream);
  run_branch(pro_x, proN, 33, proD, pro_ei, proE, pro_bat, pro_ew,  pro, sc, embxt, stream);
  run_branch(clq_x, clqN, 92, clqD, clq_ei, clqE, clq_bat, nullptr, clq, sc, embxq, stream);

  // -------- attention over branch embeddings --------
  gemm_launch(embx,  attx_l1W, attx_l1b, t64, 256, 128, 64, 2, sc.a16, sc.w16, stream);
  k_att_dot<<<grd(256L), TPB, 0, stream>>>(t64, attx_l2W, scores, 256, 0);
  gemm_launch(embxt, attt_l1W, attt_l1b, t64, 256, 128, 64, 2, sc.a16, sc.w16, stream);
  k_att_dot<<<grd(256L), TPB, 0, stream>>>(t64, attt_l2W, scores, 256, 1);
  gemm_launch(embxq, attq_l1W, attq_l1b, t64, 256, 128, 64, 2, sc.a16, sc.w16, stream);
  k_att_dot<<<grd(256L), TPB, 0, stream>>>(t64, attq_l2W, scores, 256, 2);
  k_att_emb<<<grd(256L * 128), TPB, 0, stream>>>(scores, embx, embxt, embxq, embcat, 256);

  // -------- head MLP --------
  gemm_launch(embcat, fc1W, fc1b, fc1o, 256, 384, 1024, 1, sc.a16, sc.w16, stream);
  gemm_launch(fc1o,   fc2W, fc2b, fc2o, 256, 1024, 512, 1, sc.a16, sc.w16, stream);
  gemm_launch(fc2o,   outW, outb, (float*)d_out, 256, 512, 1, 0, sc.a16, sc.w16, stream);
}
